// CNV_SNN_67130338836711
// MI455X (gfx1250) — matrix operand order repaired and checked by reading, not compiled, not run on hardware
//
#include <hip/hip_runtime.h>
#include <hip/hip_bf16.h>
#include <hip/hip_fp16.h>

// ---------------------------------------------------------------------------
// SNN (conv1 -> LIF -> conv2 -> LIF -> fc -> LIF, 25 steps) for MI455X gfx1250
//   conv2  : implicit GEMM, V_WMMA_F32_16X16X32_F16 (spikes exact in f16)
//   fc     : K-split GEMM,  V_WMMA_F32_16X16X32_F16, N padded 10 -> 16
//   conv1  : fp32 VALU with LDS-staged weights (K=25 only)
// Bandwidth-bound design: state kept in WMMA-fragment-native layouts.
// ---------------------------------------------------------------------------

typedef __attribute__((ext_vector_type(16))) _Float16 v16h;
typedef __attribute__((ext_vector_type(8)))  _Float16 v8h;
typedef __attribute__((ext_vector_type(8)))  float    v8f;

#define T_STEPS   25
#define BATCH     256
#define BETA      0.9f
#define THRESH    1.0f

// conv1 geometry
#define H1IN 28
#define W1IN 28
#define C1   16
#define H1   24
#define W1   24
#define P1   (H1*W1)            // 576 pixels per image
// conv2 geometry
#define C2   32
#define H2   20
#define W2   20
#define P2   (H2*W2)            // 400 output positions per image
#define M2TOT (BATCH*P2)        // 102400
#define M2TILES (M2TOT/16)      // 6400
#define KCHUNKS2 13             // ceil(400/32): taps 0..24 (+ zero tap 25)
// fc geometry
#define K3   (C2*P2)            // 12800
#define KCHUNKS3 (K3/32)        // 400
#define KSPLIT3  16             // 16 K-splits x 25 chunks
#define M3TILES  (BATCH/16)     // 16

// ---------------------------------------------------------------------------
__global__ __launch_bounds__(256) void zero_f32_kernel(float* __restrict__ p, int n) {
    int i = blockIdx.x * 256 + threadIdx.x;
    if (i < n) p[i] = 0.0f;
}

// ---------------------------------------------------------------------------
// Pre-shuffle conv2 weights into exact B-fragment order.
// w2frag[(c*2+nt)][lane][i] = B[K = 32c + (i<8 ? (lane>>4)*8 + i : 16 + (lane>>4)*8 + i-8)][N = nt*16 + (lane&15)]
// with K = tap*16 + ic, B[k][n] = conv2_w[oc=n][ic][tap]; tap 25 -> 0 (K pad).
__global__ __launch_bounds__(256) void prep_w2frag_kernel(
    const float* __restrict__ w2, _Float16* __restrict__ w2frag) {
    int idx = blockIdx.x * 256 + threadIdx.x;     // 26*32*16 = 13312
    if (idx >= KCHUNKS2 * 2 * 32 * 16) return;
    int i    = idx & 15;
    int lane = (idx >> 4) & 31;
    int c2   = idx >> 9;
    int c  = c2 >> 1, nt = c2 & 1;
    int K  = 32 * c + ((i < 8) ? ((lane >> 4) * 8 + i) : (16 + (lane >> 4) * 8 + (i - 8)));
    int tap = K >> 4, ic = K & 15;
    int oc = nt * 16 + (lane & 15);
    float v = (tap < 25) ? w2[(oc * 16 + ic) * 25 + tap] : 0.0f;
    w2frag[idx] = (_Float16)v;
}

// wfcfrag[c][lane][i] = fc1_w[n = lane&15][K = 32c + (i<8 ? (lane>>4)*8 + i : 16 + (lane>>4)*8 + i-8)], n>=10 -> 0
__global__ __launch_bounds__(256) void prep_wfcfrag_kernel(
    const float* __restrict__ fw, _Float16* __restrict__ wfrag) {
    int idx = blockIdx.x * 256 + threadIdx.x;     // 400*32*16 = 204800
    if (idx >= KCHUNKS3 * 32 * 16) return;
    int i    = idx & 15;
    int lane = (idx >> 4) & 31;
    int c    = idx >> 9;
    int K = 32 * c + ((i < 8) ? ((lane >> 4) * 8 + i) : (16 + (lane >> 4) * 8 + (i - 8)));
    int n = lane & 15;
    float v = (n < 10) ? fw[(size_t)n * K3 + K] : 0.0f;
    wfrag[idx] = (_Float16)v;
}

// ---------------------------------------------------------------------------
// conv1 + LIF1.  One thread per output pixel, all 16 channels.
// s1 written channel-innermost f16: s1[(b*576 + y*24 + x)*16 + c]
__global__ __launch_bounds__(256) void conv1_lif_kernel(
    const float* __restrict__ xt,     // [B][28][28]
    const float* __restrict__ w1,     // [16][5][5]
    const float* __restrict__ b1,     // [16]
    float*       __restrict__ mem1,   // [B*576][16]
    _Float16*    __restrict__ s1) {   // [B*576][16] (+pad)
    __shared__ float wsm[25 * 16];    // wsm[tap*16 + oc]
    for (int j = threadIdx.x; j < 400; j += 256) {
        int tap = j >> 4, oc = j & 15;
        wsm[j] = w1[oc * 25 + tap];
    }
    __syncthreads();

    int idx = blockIdx.x * 256 + threadIdx.x;     // exactly B*576 threads
    int b  = idx / P1, p = idx % P1;
    int oy = p / W1,  ox = p % W1;

    float acc[16];
#pragma unroll
    for (int oc = 0; oc < 16; ++oc) acc[oc] = b1[oc];

    const float* xb = xt + (size_t)b * (H1IN * W1IN) + oy * W1IN + ox;
#pragma unroll
    for (int ky = 0; ky < 5; ++ky) {
#pragma unroll
        for (int kx = 0; kx < 5; ++kx) {
            float xv = xb[ky * W1IN + kx];
            const float* wr = &wsm[(ky * 5 + kx) * 16];
#pragma unroll
            for (int oc = 0; oc < 16; ++oc) acc[oc] = fmaf(xv, wr[oc], acc[oc]);
        }
    }

    float*    mp = mem1 + (size_t)idx * 16;
    _Float16* sp = s1   + (size_t)idx * 16;
#pragma unroll
    for (int oc = 0; oc < 16; ++oc) {
        float mo = mp[oc];
        float mn = BETA * mo + acc[oc] - ((mo > THRESH) ? THRESH : 0.0f);
        mp[oc] = mn;
        sp[oc] = (_Float16)((mn > THRESH) ? 1.0f : 0.0f);
    }
}

// ---------------------------------------------------------------------------
// conv2 as implicit GEMM (M=102400, N=32, K=400->416) + LIF2.
// One wave per 16-row M-tile; 2 WMMAs (N-tiles) per 32-deep K chunk.
// mem2 kept in C/D-fragment layout: mem2[(tile*2+nt)*32 + lane][r] (8 f32).
__global__ __launch_bounds__(256) void conv2_wmma_lif_kernel(
    const _Float16* __restrict__ s1,      // [B*576][16] (+ zero pad)
    const _Float16* __restrict__ w2frag,  // [26][32][16]
    const float*    __restrict__ b2,      // [32]
    float*          __restrict__ mem2,    // [6400*2*32][8]
    _Float16*       __restrict__ s2fc) {  // [B][12800]  k = oc*400 + p
    int wid  = blockIdx.x * 8 + (threadIdx.x >> 5);  // wave-uniform
    if (wid >= M2TILES) return;                       // whole-wave exit
    int lane = threadIdx.x & 31;
    int hi   = lane >> 4, lo = lane & 15;

    // A-fragment row for this lane
    int m  = wid * 16 + lo;
    int b  = m / P2, p = m % P2;
    int oy = p / W2, ox = p % W2;
    const _Float16* aptr =
        s1 + ((size_t)(b * P1 + oy * W1 + ox) * 16 + hi * 8);
    const _Float16* bptr = w2frag + (size_t)lane * 16;

    v8f acc0 = {}, acc1 = {};
    for (int c = 0; c < KCHUNKS2; ++c) {
        int t0 = 2 * c, t1 = 2 * c + 1;
        // taps: ky = t/5, kx = t%5 ; channel-inner layout -> contiguous 16B
        v8h a0 = *(const v8h*)(aptr + ((t0 / 5) * W1 + (t0 % 5)) * 16);
        v8h a1 = *(const v8h*)(aptr + ((t1 / 5) * W1 + (t1 % 5)) * 16);
        v16h a = __builtin_shufflevector(a0, a1, 0, 1, 2, 3, 4, 5, 6, 7,
                                                 8, 9, 10, 11, 12, 13, 14, 15);
        v16h bf0 = *(const v16h*)(bptr + (size_t)(c * 2 + 0) * 32 * 16);
        v16h bf1 = *(const v16h*)(bptr + (size_t)(c * 2 + 1) * 32 * 16);
        __builtin_prefetch(bptr + (size_t)(c * 2 + 2) * 32 * 16, 0, 3);
        acc0 = __builtin_amdgcn_wmma_f32_16x16x32_f16(
            false, a, false, bf0, (short)0, acc0, false, false);
        acc1 = __builtin_amdgcn_wmma_f32_16x16x32_f16(
            false, a, false, bf1, (short)0, acc1, false, false);
    }

    // LIF2 epilogue: C/D layout lane holds (M = tile*16 + r + 8*hi, N = nt*16+lo)
#pragma unroll
    for (int nt = 0; nt < 2; ++nt) {
        v8f    acc  = nt ? acc1 : acc0;
        float  bias = b2[nt * 16 + lo];
        float* mp   = mem2 + (((size_t)wid * 2 + nt) * 32 + lane) * 8;
        int    oc   = nt * 16 + lo;
#pragma unroll
        for (int r = 0; r < 8; ++r) {
            int mr = wid * 16 + r + 8 * hi;
            int br = mr / P2, pr = mr % P2;
            float mo  = mp[r];
            float cur = acc[r] + bias;
            float mn  = BETA * mo + cur - ((mo > THRESH) ? THRESH : 0.0f);
            mp[r] = mn;
            s2fc[(size_t)br * K3 + oc * P2 + pr] =
                (_Float16)((mn > THRESH) ? 1.0f : 0.0f);
        }
    }
}

// ---------------------------------------------------------------------------
// fc: M=256, N=16(pad), K=12800 ; 16 M-tiles x 16 K-splits, f32 atomics reduce.
__global__ __launch_bounds__(256) void fc_wmma_kernel(
    const _Float16* __restrict__ s2fc,    // [256][12800]
    const _Float16* __restrict__ wfrag,   // [400][32][16]
    float*          __restrict__ cur3acc) // [256][16]
{
    int wid  = blockIdx.x * 8 + (threadIdx.x >> 5);  // 0..255
    int mt   = wid & (M3TILES - 1);
    int ks   = wid >> 4;
    int lane = threadIdx.x & 31;
    int hi   = lane >> 4, lo = lane & 15;

    int brow = mt * 16 + lo;
    const _Float16* ap = s2fc + (size_t)brow * K3 + hi * 8;

    v8f acc = {};
    int c0 = ks * (KCHUNKS3 / KSPLIT3);
    for (int c = c0; c < c0 + (KCHUNKS3 / KSPLIT3); ++c) {
        v8h a0 = *(const v8h*)(ap + 32 * c);
        v8h a1 = *(const v8h*)(ap + 32 * c + 16);
        __builtin_prefetch(ap + 32 * (c + 1), 0, 3);
        v16h a = __builtin_shufflevector(a0, a1, 0, 1, 2, 3, 4, 5, 6, 7,
                                                 8, 9, 10, 11, 12, 13, 14, 15);
        v16h bf = *(const v16h*)(wfrag + ((size_t)c * 32 + lane) * 16);
        acc = __builtin_amdgcn_wmma_f32_16x16x32_f16(
            false, a, false, bf, (short)0, acc, false, false);
    }
#pragma unroll
    for (int r = 0; r < 8; ++r) {
        int br = mt * 16 + r + 8 * hi;
        __hip_atomic_fetch_add(&cur3acc[br * 16 + lo], acc[r],
                               __ATOMIC_RELAXED, __HIP_MEMORY_SCOPE_AGENT);
    }
}

// ---------------------------------------------------------------------------
// LIF3 + spike-count accumulation into d_out; resets cur3acc for next step.
__global__ __launch_bounds__(256) void lif3_kernel(
    float* __restrict__ mem3, float* __restrict__ cur3acc,
    const float* __restrict__ fcb, float* __restrict__ out) {
    int t = blockIdx.x * 256 + threadIdx.x;   // 4096
    if (t >= BATCH * 16) return;
    int b = t >> 4, n = t & 15;
    float cur = cur3acc[t] + ((n < 10) ? fcb[n] : 0.0f);
    cur3acc[t] = 0.0f;
    float mo = mem3[t];
    float mn = BETA * mo + cur - ((mo > THRESH) ? THRESH : 0.0f);
    mem3[t] = mn;
    if (n < 10 && mn > THRESH) out[b * 10 + n] += 1.0f;
}

// ---------------------------------------------------------------------------
extern "C" void kernel_launch(void* const* d_in, const int* in_sizes, int n_in,
                              void* d_out, int out_size, void* d_ws, size_t ws_size,
                              hipStream_t stream) {
    const float* x   = (const float*)d_in[0];   // [25][256][1][28][28]
    const float* w1  = (const float*)d_in[1];   // [16][1][5][5]
    const float* b1  = (const float*)d_in[2];   // [16]
    const float* w2  = (const float*)d_in[3];   // [32][16][5][5]
    const float* b2  = (const float*)d_in[4];   // [32]
    const float* fw  = (const float*)d_in[5];   // [10][12800]
    const float* fb  = (const float*)d_in[6];   // [10]
    // d_in[7] = num_steps (device scalar) — reference fixes it at 25.
    float* out = (float*)d_out;                 // [256][10]

    // -------- workspace carve-up (256B aligned slices) --------
    char*  ws  = (char*)d_ws;
    size_t off = 0;
    auto carve = [&](size_t bytes) -> void* {
        void* p = ws + off;
        off = (off + bytes + 255) & ~(size_t)255;
        return p;
    };
    const size_t S1_ELEMS = (size_t)BATCH * P1 * 16 + 1024;  // + OOB pad (tap 25)
    _Float16* s1      = (_Float16*)carve(S1_ELEMS * sizeof(_Float16));
    float*    mem1    = (float*)   carve((size_t)BATCH * P1 * 16 * sizeof(float));
    float*    mem2    = (float*)   carve((size_t)M2TILES * 2 * 32 * 8 * sizeof(float));
    _Float16* s2fc    = (_Float16*)carve((size_t)BATCH * K3 * sizeof(_Float16));
    _Float16* w2frag  = (_Float16*)carve((size_t)KCHUNKS2 * 2 * 32 * 16 * sizeof(_Float16));
    _Float16* wfcfrag = (_Float16*)carve((size_t)KCHUNKS3 * 32 * 16 * sizeof(_Float16));
    float*    mem3    = (float*)   carve((size_t)BATCH * 16 * sizeof(float));
    float*    cur3acc = (float*)   carve((size_t)BATCH * 16 * sizeof(float));

    auto zero = [&](float* p, size_t n) {
        zero_f32_kernel<<<(unsigned)((n + 255) / 256), 256, 0, stream>>>(p, (int)n);
    };
    // deterministic per-call state init (ws/d_out contents are otherwise stale)
    zero((float*)s1, S1_ELEMS / 2);                 // zeros f16 spikes + pad
    zero(mem1, (size_t)BATCH * P1 * 16);
    zero(mem2, (size_t)M2TILES * 2 * 32 * 8);
    zero(mem3, (size_t)BATCH * 16);
    zero(cur3acc, (size_t)BATCH * 16);
    zero(out, (size_t)BATCH * 10);

    prep_w2frag_kernel<<<(KCHUNKS2 * 2 * 32 * 16 + 255) / 256, 256, 0, stream>>>(w2, w2frag);
    prep_wfcfrag_kernel<<<(KCHUNKS3 * 32 * 16 + 255) / 256, 256, 0, stream>>>(fw, wfcfrag);

    for (int t = 0; t < T_STEPS; ++t) {
        const float* xt = x + (size_t)t * BATCH * H1IN * W1IN;
        conv1_lif_kernel<<<(BATCH * P1) / 256, 256, 0, stream>>>(xt, w1, b1, mem1, s1);
        conv2_wmma_lif_kernel<<<M2TILES / 8, 256, 0, stream>>>(s1, w2frag, b2, mem2, s2fc);
        fc_wmma_kernel<<<(M3TILES * KSPLIT3) / 8, 256, 0, stream>>>(s2fc, wfcfrag, cur3acc);
        lif3_kernel<<<(BATCH * 16) / 256, 256, 0, stream>>>(mem3, cur3acc, fb, out);
    }
}
